// RNN_language_model_22084721836274
// MI455X (gfx1250) — matrix operand order repaired and checked by reading, not compiled, not run on hardware
//
#include <hip/hip_runtime.h>
#include <hip/hip_bf16.h>
#include <math.h>

// ---------------- problem constants ----------------
#define HH    512      // hidden
#define BB    64       // batch
#define TT1   128      // timesteps (T-1)
#define VV    8001     // vocab
#define VPADC 8064     // vocab padded to multiple of 64 (decoder macro-tile width)
#define G4    2048     // 4*H
#define EPSBN 1e-5f

typedef __attribute__((ext_vector_type(16))) __bf16 v16bf;
typedef __attribute__((ext_vector_type(8)))  float  v8f;

// ---------------- WMMA fragment loaders ----------------
// 16-bit A-matrix 16x32 layout (ISA 7.12.2): lane = half*16 + m,
// elements 0..7 -> K = half*8 + e ; elements 8..15 -> K = 16 + half*8 + (e-8)
__device__ __forceinline__ v16bf frag_a(const __bf16* rowk0, int half) {
  union { v16bf v; float4 f[2]; } u;
  u.f[0] = *(const float4*)(rowk0 + half * 8);
  u.f[1] = *(const float4*)(rowk0 + 16 + half * 8);
  return u.v;
}
// 16-bit B-matrix 32x16 layout: lane = half*16 + n,
// elements 0..7 -> K = half*8 + e ; elements 8..15 -> K = 16 + half*8 + (e-8)
__device__ __forceinline__ v16bf frag_b(const __bf16* rowk0, int half) {
  union { v16bf v; float4 f[2]; } u;
  u.f[0] = *(const float4*)(rowk0 + half * 8);
  u.f[1] = *(const float4*)(rowk0 + 16 + half * 8);
  return u.v;
}

// ---------------- small utility kernels ----------------
__global__ void k_f2bf(const float* __restrict__ s, __bf16* __restrict__ d, int n) {
  int i = blockIdx.x * blockDim.x + threadIdx.x;
  if (i < n) d[i] = (__bf16)s[i];
}
__global__ void k_f2bf_pad(const float* __restrict__ s, __bf16* __restrict__ d, int n, int npad) {
  int i = blockIdx.x * blockDim.x + threadIdx.x;
  if (i < npad) d[i] = (i < n) ? (__bf16)s[i] : (__bf16)0.0f;
}
__global__ void k_addbias(const float* __restrict__ a, const float* __restrict__ b,
                          float* __restrict__ o, int n) {
  int i = blockIdx.x * blockDim.x + threadIdx.x;
  if (i < n) o[i] = a[i] + b[i];
}
__global__ void k_zero_state(float* c1, float* c2, __bf16* h1bf, __bf16* h2bf) {
  int i = blockIdx.x * blockDim.x + threadIdx.x;   // 32768 threads
  c1[i] = 0.f; c2[i] = 0.f;
  h1bf[i] = (__bf16)0.0f; h2bf[i] = (__bf16)0.0f;
}

// ---------------- big GEMM: XG1 = emb[x] @ W_ih1^T + (b_ih1+b_hh1) ----------------
// M = 8192 rows (m = t*64 + b), N = 2048, K = 512. Wave tile 32x32 (2x2 subtiles).
__global__ void k_gemm_xg1(const int* __restrict__ x, const __bf16* __restrict__ embbf,
                           const __bf16* __restrict__ wbf, const float* __restrict__ bias,
                           float* __restrict__ xg1) {
  int wt = blockIdx.x * 8 + (threadIdx.x >> 5);  // 256 x 64 = 16384 wave tiles
  int mt = wt >> 6;           // 0..255
  int nt = wt & 63;           // 0..63
  int lane = threadIdx.x & 31;
  int half = lane >> 4;
  int m0 = mt * 32, n0 = nt * 32;
  // 32-row tile stays within one t (64 rows per t, m0 multiple of 32)
  int t = m0 >> 6;
  int b0 = (m0 & 63) + (lane & 15);
  int tok0 = x[b0 * 129 + t];
  int tok1 = x[(b0 + 16) * 129 + t];
  const __bf16* arow0 = embbf + (size_t)tok0 * HH;
  const __bf16* arow1 = embbf + (size_t)tok1 * HH;
  const __bf16* wrow0 = wbf + (size_t)(n0 + (lane & 15)) * HH;
  const __bf16* wrow1 = wrow0 + 16 * HH;
  v8f acc[2][2];
#pragma unroll
  for (int i = 0; i < 2; ++i)
#pragma unroll
    for (int j = 0; j < 2; ++j) { v8f z = {}; acc[i][j] = z; }
#pragma unroll 4
  for (int k0 = 0; k0 < HH; k0 += 32) {
    v16bf a0 = frag_a(arow0 + k0, half);
    v16bf a1 = frag_a(arow1 + k0, half);
    v16bf bf0 = frag_b(wrow0 + k0, half);
    v16bf bf1 = frag_b(wrow1 + k0, half);
    acc[0][0] = __builtin_amdgcn_wmma_f32_16x16x32_bf16(false, a0, false, bf0, (short)0, acc[0][0], false, false);
    acc[1][0] = __builtin_amdgcn_wmma_f32_16x16x32_bf16(false, a1, false, bf0, (short)0, acc[1][0], false, false);
    acc[0][1] = __builtin_amdgcn_wmma_f32_16x16x32_bf16(false, a0, false, bf1, (short)0, acc[0][1], false, false);
    acc[1][1] = __builtin_amdgcn_wmma_f32_16x16x32_bf16(false, a1, false, bf1, (short)0, acc[1][1], false, false);
  }
#pragma unroll
  for (int j = 0; j < 2; ++j) {
    int n = n0 + j * 16 + (lane & 15);
    float bi = bias[n];
#pragma unroll
    for (int i = 0; i < 2; ++i) {
#pragma unroll
      for (int r = 0; r < 8; ++r) {
        int m = m0 + i * 16 + half * 8 + r;
        xg1[(size_t)m * G4 + n] = acc[i][j][r] + bi;
      }
    }
  }
}

// ---------------- per-step GEMM 1: gates = XG1[t] + h1 @ W_hh1^T ----------------
// M = 64, N = 2048, K = 512. One 16x16 tile per wave (maximize WGP spread: latency path).
__global__ void k_gates1(const float* __restrict__ xg_t, const __bf16* __restrict__ hbf,
                         const __bf16* __restrict__ wbf, float* __restrict__ gates) {
  int tile = blockIdx.x * 8 + (threadIdx.x >> 5);  // 512 tiles
  int mt = tile >> 7;          // 0..3
  int nt = tile & 127;
  int lane = threadIdx.x & 31;
  int half = lane >> 4;
  int m0 = mt * 16, n0 = nt * 16;
  const __bf16* arow = hbf + (size_t)(m0 + (lane & 15)) * HH;
  const __bf16* wrow = wbf + (size_t)(n0 + (lane & 15)) * HH;
  v8f acc = {};
#pragma unroll 4
  for (int k0 = 0; k0 < HH; k0 += 32) {
    v16bf af = frag_a(arow + k0, half);
    v16bf bf = frag_b(wrow + k0, half);
    acc = __builtin_amdgcn_wmma_f32_16x16x32_bf16(false, af, false, bf, (short)0, acc, false, false);
  }
  int n = n0 + (lane & 15);
#pragma unroll
  for (int r = 0; r < 8; ++r) {
    int m = m0 + half * 8 + r;
    gates[(size_t)m * G4 + n] = acc[r] + xg_t[(size_t)m * G4 + n];
  }
}

// ---------------- per-step GEMM 2: gates = a@W_ih2^T + h2@W_hh2^T + bias2 ----------------
__global__ void k_gates2(const __bf16* __restrict__ abf, const __bf16* __restrict__ wibf,
                         const __bf16* __restrict__ hbf, const __bf16* __restrict__ whbf,
                         const float* __restrict__ bias, float* __restrict__ gates) {
  int tile = blockIdx.x * 8 + (threadIdx.x >> 5);
  int mt = tile >> 7;
  int nt = tile & 127;
  int lane = threadIdx.x & 31;
  int half = lane >> 4;
  int m0 = mt * 16, n0 = nt * 16;
  const __bf16* arow1 = abf + (size_t)(m0 + (lane & 15)) * HH;
  const __bf16* arow2 = hbf + (size_t)(m0 + (lane & 15)) * HH;
  const __bf16* wrow1 = wibf + (size_t)(n0 + (lane & 15)) * HH;
  const __bf16* wrow2 = whbf + (size_t)(n0 + (lane & 15)) * HH;
  v8f acc = {};
#pragma unroll 4
  for (int k0 = 0; k0 < HH; k0 += 32) {
    v16bf af = frag_a(arow1 + k0, half);
    v16bf bf = frag_b(wrow1 + k0, half);
    acc = __builtin_amdgcn_wmma_f32_16x16x32_bf16(false, af, false, bf, (short)0, acc, false, false);
    v16bf af2 = frag_a(arow2 + k0, half);
    v16bf bf2 = frag_b(wrow2 + k0, half);
    acc = __builtin_amdgcn_wmma_f32_16x16x32_bf16(false, af2, false, bf2, (short)0, acc, false, false);
  }
  int n = n0 + (lane & 15);
  float bi = bias[n];
#pragma unroll
  for (int r = 0; r < 8; ++r) {
    int m = m0 + half * 8 + r;
    gates[(size_t)m * G4 + n] = acc[r] + bi;
  }
}

// ---------------- fused LSTM cell + batch-norm + mask (single workgroup) ----------------
// Thread j owns column j: cell update for all 64 batch rows (h kept in registers),
// biased batch stats, then writes bf16 recurrent state hbf and bf16 normalized output.
__global__ void k_cellbn(const float* __restrict__ gates, float* __restrict__ c,
                         const float* __restrict__ gamma, const float* __restrict__ beta,
                         const float* __restrict__ mask,
                         __bf16* __restrict__ hbf, __bf16* __restrict__ outbf, int ostride) {
  int j = threadIdx.x;                 // 512 threads, 1 block
  float hcol[BB];
  float s = 0.f, s2 = 0.f;
#pragma unroll
  for (int b = 0; b < BB; ++b) {
    const float* g = gates + (size_t)b * G4;
    float gi = g[j], gf = g[512 + j], gg = g[1024 + j], go = g[1536 + j];
    float si = 1.f / (1.f + __expf(-gi));
    float sf = 1.f / (1.f + __expf(-gf));
    float so = 1.f / (1.f + __expf(-go));
    float cn = sf * c[b * HH + j] + si * tanhf(gg);
    float hn = so * tanhf(cn);
    c[b * HH + j] = cn;
    hbf[b * HH + j] = (__bf16)hn;
    hcol[b] = hn;
    s += hn; s2 += hn * hn;
  }
  float mu = s * (1.f / BB);
  float var = s2 * (1.f / BB) - mu * mu;
  float alpha = gamma[j] * rsqrtf(var + EPSBN);
  float bet = beta[j] - mu * alpha;
#pragma unroll
  for (int b = 0; b < BB; ++b) {
    float v = hcol[b] * alpha + bet;
    outbf[(size_t)b * ostride + j] = (__bf16)(v * mask[b * HH + j]);
  }
}

// ---------------- big decoder GEMM: out[b,v,t] = Bact[b,t,:] . Wdec[v,:] + bdec[v] ----------
// Bact rows m = b*128 + t. Wave tile 32(m) x 64(n): 2 A-frags x 4 B-frags -> 8 WMMA / 12 loads
// (~21 FLOP/byte from L2). 32-row M-subtile stays inside one b -> contiguous float4 stores on t.
// M = 8192, N = VPADC (8064, zero-padded), K = 512.
__global__ void k_dec(const __bf16* __restrict__ bact, const __bf16* __restrict__ wdecbf,
                      const float* __restrict__ bdec, float* __restrict__ out) {
  int wt = blockIdx.x * 8 + (threadIdx.x >> 5);  // 256 x 126 = 32256 wave tiles
  int mt = wt / 126;          // 0..255
  int nt = wt - mt * 126;     // 0..125
  int lane = threadIdx.x & 31;
  int half = lane >> 4;
  int m0 = mt * 32, n0 = nt * 64;
  const __bf16* arow0 = bact + (size_t)(m0 + (lane & 15)) * HH;
  const __bf16* arow1 = arow0 + 16 * HH;
  const __bf16* wrow = wdecbf + (size_t)(n0 + (lane & 15)) * HH;
  v8f acc[2][4];
#pragma unroll
  for (int i = 0; i < 2; ++i)
#pragma unroll
    for (int j = 0; j < 4; ++j) { v8f z = {}; acc[i][j] = z; }
#pragma unroll 2
  for (int k0 = 0; k0 < HH; k0 += 32) {
    v16bf a0 = frag_a(arow0 + k0, half);
    v16bf a1 = frag_a(arow1 + k0, half);
#pragma unroll
    for (int j = 0; j < 4; ++j) {
      v16bf bf = frag_b(wrow + (size_t)j * 16 * HH + k0, half);
      acc[0][j] = __builtin_amdgcn_wmma_f32_16x16x32_bf16(false, a0, false, bf, (short)0, acc[0][j], false, false);
      acc[1][j] = __builtin_amdgcn_wmma_f32_16x16x32_bf16(false, a1, false, bf, (short)0, acc[1][j], false, false);
    }
  }
  int b = m0 >> 7;                       // fixed within tile
#pragma unroll
  for (int j = 0; j < 4; ++j) {
    int n = n0 + j * 16 + (lane & 15);
    if (n < VV) {
      float bi = bdec[n];
      float* pbase = out + ((size_t)b * VV + n) * TT1;
#pragma unroll
      for (int i = 0; i < 2; ++i) {
        int t0 = (m0 & 127) + i * 16 + half * 8;
        float4 s0 = make_float4(acc[i][j][0] + bi, acc[i][j][1] + bi,
                                acc[i][j][2] + bi, acc[i][j][3] + bi);
        float4 s1 = make_float4(acc[i][j][4] + bi, acc[i][j][5] + bi,
                                acc[i][j][6] + bi, acc[i][j][7] + bi);
        *(float4*)(pbase + t0) = s0;
        *(float4*)(pbase + t0 + 4) = s1;
      }
    }
  }
}

// ---------------- host driver ----------------
extern "C" void kernel_launch(void* const* d_in, const int* in_sizes, int n_in,
                              void* d_out, int out_size, void* d_ws, size_t ws_size,
                              hipStream_t stream) {
  (void)in_sizes; (void)n_in; (void)out_size; (void)ws_size;
  const int*   x      = (const int*)d_in[0];
  const float* emb    = (const float*)d_in[1];
  const float* Wih1   = (const float*)d_in[2];
  const float* Whh1   = (const float*)d_in[3];
  const float* bih1   = (const float*)d_in[4];
  const float* bhh1   = (const float*)d_in[5];
  const float* gamma1 = (const float*)d_in[6];
  const float* beta1  = (const float*)d_in[7];
  const float* mask1  = (const float*)d_in[8];
  const float* Wih2   = (const float*)d_in[9];
  const float* Whh2   = (const float*)d_in[10];
  const float* bih2   = (const float*)d_in[11];
  const float* bhh2   = (const float*)d_in[12];
  const float* gamma2 = (const float*)d_in[13];
  const float* beta2  = (const float*)d_in[14];
  const float* mask2  = (const float*)d_in[15];
  const float* Wdec   = (const float*)d_in[16];
  const float* bdec   = (const float*)d_in[17];
  float* out = (float*)d_out;

  char* ws = (char*)d_ws;
  size_t off = 0;
  auto alloc = [&](size_t bytes) -> void* {
    void* p = ws + off;
    off = (off + bytes + 255) & ~(size_t)255;
    return p;
  };

  __bf16* embbf  = (__bf16*)alloc((size_t)VV * HH * 2);
  __bf16* wih1bf = (__bf16*)alloc((size_t)G4 * HH * 2);
  __bf16* whh1bf = (__bf16*)alloc((size_t)G4 * HH * 2);
  __bf16* wih2bf = (__bf16*)alloc((size_t)G4 * HH * 2);
  __bf16* whh2bf = (__bf16*)alloc((size_t)G4 * HH * 2);
  __bf16* wdecbf = (__bf16*)alloc((size_t)VPADC * HH * 2);
  float*  bias1  = (float*)alloc((size_t)G4 * 4);
  float*  bias2  = (float*)alloc((size_t)G4 * 4);
  float*  xg1    = (float*)alloc((size_t)TT1 * BB * G4 * 4);   // 67 MB
  float*  gates1 = (float*)alloc((size_t)BB * G4 * 4);
  float*  gates2 = (float*)alloc((size_t)BB * G4 * 4);
  float*  c1     = (float*)alloc((size_t)BB * HH * 4);
  float*  c2     = (float*)alloc((size_t)BB * HH * 4);
  __bf16* h1bf   = (__bf16*)alloc((size_t)BB * HH * 2);
  __bf16* h2bf   = (__bf16*)alloc((size_t)BB * HH * 2);
  __bf16* abf    = (__bf16*)alloc((size_t)BB * HH * 2);
  __bf16* bactbf = (__bf16*)alloc((size_t)BB * TT1 * HH * 2);  // rows m = b*128 + t

  const int TPB = 256;
  // weight / embedding conversion to bf16 (once per launch; deterministic)
  int nEmb = VV * HH;
  k_f2bf<<<(nEmb + TPB - 1) / TPB, TPB, 0, stream>>>(emb, embbf, nEmb);
  int nW = G4 * HH;
  k_f2bf<<<(nW + TPB - 1) / TPB, TPB, 0, stream>>>(Wih1, wih1bf, nW);
  k_f2bf<<<(nW + TPB - 1) / TPB, TPB, 0, stream>>>(Whh1, whh1bf, nW);
  k_f2bf<<<(nW + TPB - 1) / TPB, TPB, 0, stream>>>(Wih2, wih2bf, nW);
  k_f2bf<<<(nW + TPB - 1) / TPB, TPB, 0, stream>>>(Whh2, whh2bf, nW);
  int nDecPad = VPADC * HH;
  k_f2bf_pad<<<(nDecPad + TPB - 1) / TPB, TPB, 0, stream>>>(Wdec, wdecbf, nEmb, nDecPad);
  k_addbias<<<(G4 + TPB - 1) / TPB, TPB, 0, stream>>>(bih1, bhh1, bias1, G4);
  k_addbias<<<(G4 + TPB - 1) / TPB, TPB, 0, stream>>>(bih2, bhh2, bias2, G4);
  k_zero_state<<<(BB * HH) / TPB, TPB, 0, stream>>>(c1, c2, h1bf, h2bf);

  // Hoisted layer-1 input projection for all 128 steps: [8192x512] x [512x2048], 32x32/wave
  k_gemm_xg1<<<(256 * 64) / 8, TPB, 0, stream>>>(x, embbf, wih1bf, bias1, xg1);

  // Sequential recurrence (critical path): 4 launches per step
  for (int t = 0; t < TT1; ++t) {
    k_gates1<<<(4 * 128) / 8, TPB, 0, stream>>>(xg1 + (size_t)t * BB * G4, h1bf, whh1bf, gates1);
    k_cellbn<<<1, HH, 0, stream>>>(gates1, c1, gamma1, beta1, mask1, h1bf, abf, HH);
    k_gates2<<<(4 * 128) / 8, TPB, 0, stream>>>(abf, wih2bf, h2bf, whh2bf, bias2, gates2);
    // store as rows m = b*128 + t -> base offset t*HH, row stride TT1*HH
    k_cellbn<<<1, HH, 0, stream>>>(gates2, c2, gamma2, beta2, mask2, h2bf,
                                   bactbf + (size_t)t * HH, TT1 * HH);
  }

  // Hoisted decoder for all 128 steps: [8192x512] x [512x8064], 32x64/wave,
  // contiguous t-major float4 stores into out[b,v,t]
  k_dec<<<(256 * 126) / 8, TPB, 0, stream>>>(bactbf, wdecbf, bdec, out);
}
